// PhysicsTransformer_20066087207296
// MI455X (gfx1250) — matrix operand order repaired and checked by reading, not compiled, not run on hardware
//
#include <hip/hip_runtime.h>
#include <hip/hip_bf16.h>

#define HID   128
#define NRBF  32
#define NHEAD 4
#define HDIM  32
#define NLAYER 5
#define PI_F  3.14159265358979f
#define CUTOFF_F 2.5f

typedef __attribute__((ext_vector_type(16))) __bf16 bfrag16;  // A/B frag: 16 bf16 = 8 VGPRs
typedef __attribute__((ext_vector_type(8)))  float  v8f;      // C/D frag: 8 f32 VGPRs
typedef __attribute__((ext_vector_type(4))) unsigned int u32x4;
typedef __attribute__((ext_vector_type(8))) int i32x8;
typedef __attribute__((ext_vector_type(4))) int i32x4;

union ABu { uint4 u4[2]; bfrag16 f; };

// GEMM epilogue mode bits (compile-time)
#define GM_SILU 1
#define GM_ACC  2
#define GM_WF32 4
#define GM_WB16 8
#define GM_RS   16

#if defined(__has_builtin)
#if __has_builtin(__builtin_amdgcn_tensor_load_to_lds) && __has_builtin(__builtin_amdgcn_s_wait_tensorcnt)
#define HAVE_TDM 1
#endif
#endif

__device__ inline unsigned short f2bf(float f) {
    unsigned u = __float_as_uint(f);
    u += 0x7fffu + ((u >> 16) & 1u);      // round-to-nearest-even
    return (unsigned short)(u >> 16);
}
__device__ inline unsigned pk2bf(float lo, float hi) {
    return (unsigned)f2bf(lo) | ((unsigned)f2bf(hi) << 16);
}
__device__ inline float bf2f(unsigned short b) {
    return __uint_as_float(((unsigned)b) << 16);
}
__device__ inline void up2(unsigned u, float& a, float& b) {
    a = bf2f((unsigned short)(u & 0xffffu));
    b = bf2f((unsigned short)(u >> 16));
}
__device__ inline float silu(float x) {
    return x * __builtin_amdgcn_rcpf(1.0f + __expf(-x));  // v_rcp_f32, no IEEE div
}

#ifdef HAVE_TDM
// ---------------------------------------------------------------------------
// Tensor Data Mover: 1-D DMA of `bytes` (multiple of 4) from global -> LDS.
// D# per CDNA5 ISA: count=1, type=2 ("image"),
// data_size=4B (code 2), tile_dim0 = tensor_dim0 = stride = dwords,
// tile_dim1 = tensor_dim1 = 1. Groups 2/3 zero. 6-arg builtin (clang-23).
// Issue from ONE wave only (EXEC ignored: one DMA per issuing wave).
// ---------------------------------------------------------------------------
__device__ inline void tdm_load_to_lds(const void* gptr, unsigned lds_off, unsigned bytes) {
    unsigned dwords = bytes >> 2;
    unsigned long long ga = (unsigned long long)gptr;
    u32x4 g0;
    g0[0] = 1u;                                    // [1:0] count=1, user mode
    g0[1] = lds_off;                               // [63:32] lds_addr
    g0[2] = (unsigned)ga;                          // global_addr[31:0]
    g0[3] = (unsigned)(ga >> 32) | (2u << 30);     // global_addr[56:32] | type=2
    i32x8 g1;
    g1[0] = (int)(2u << 16);                       // data_size = 4B
    g1[1] = (int)((dwords & 0xffffu) << 16);       // tensor_dim0[15:0]
    g1[2] = (int)(((dwords >> 16) & 0xffffu) | (1u << 16)); // dim0[31:16], tensor_dim1=1
    g1[3] = (int)((dwords & 0xffffu) << 16);       // tile_dim0 = dwords (<=65535)
    g1[4] = 1;                                     // tile_dim1 = 1
    g1[5] = (int)dwords;                           // tensor_dim0_stride[31:0]
    g1[6] = 0;
    g1[7] = 0;
    i32x4 z4 = {0, 0, 0, 0};
    i32x8 z8 = {0, 0, 0, 0, 0, 0, 0, 0};
    __builtin_amdgcn_tensor_load_to_lds(g0, g1, z4, z4, z8, 0);
}
#endif

// ---------------------------------------------------------------------------
// Weight prep: W[K][Nc] f32  ->  Wt[Nc][K] bf16  (row of Wt = column of W)
// ---------------------------------------------------------------------------
__global__ void k_tw(const float* __restrict__ W, unsigned short* __restrict__ Wt,
                     int K, int Nc) {
    int idx = blockIdx.x * 256 + threadIdx.x;
    if (idx >= K * Nc) return;
    int k = idx / Nc, n = idx - k * Nc;
    Wt[(size_t)n * K + k] = f2bf(W[idx]);
}

// ---------------------------------------------------------------------------
// Per-edge Bessel RBF (bf16, [E,32] row-major) + cosine cutoff (f32)
// ---------------------------------------------------------------------------
__global__ void k_edge_rbf(const float* __restrict__ ea,
                           unsigned short* __restrict__ rbf_bf,
                           float* __restrict__ cut, int E) {
    int e = blockIdx.x * 256 + threadIdx.x;
    if (e >= E) return;
    float d  = ea[e];
    float dn = d * (1.0f / CUTOFF_F);
    float inv = __builtin_amdgcn_rcpf(dn + 1e-6f);
    cut[e] = 0.5f * (__cosf(d * (PI_F / CUTOFF_F)) + 1.0f) * (d < CUTOFF_F ? 1.0f : 0.0f);
    uint4 out[4];
    unsigned short* p = (unsigned short*)out;
    #pragma unroll
    for (int i = 0; i < NRBF; ++i)
        p[i] = f2bf(__sinf((float)(i + 1) * PI_F * dn) * inv);
    uint4* dst = (uint4*)(rbf_bf + (size_t)e * NRBF);
    #pragma unroll
    for (int i = 0; i < 4; ++i) dst[i] = out[i];
}

// ---------------------------------------------------------------------------
// Embed stage 1: t = silu(LN(x@e_w1 + e_b1))  -> bf16 [N,128]
// ---------------------------------------------------------------------------
__global__ void k_embed(const float* __restrict__ x, const float* __restrict__ w1,
                        const float* __restrict__ b1, const float* __restrict__ g,
                        const float* __restrict__ be, unsigned short* __restrict__ t_bf,
                        int N) {
    int n = blockIdx.x;
    int c = threadIdx.x;
    if (n >= N) return;
    float xv[5];
    #pragma unroll
    for (int i = 0; i < 5; ++i) xv[i] = x[(size_t)n * 5 + i];
    float v = b1[c];
    #pragma unroll
    for (int i = 0; i < 5; ++i) v += xv[i] * w1[i * HID + c];

    __shared__ float red[4];
    int wid = c >> 5, ln = c & 31;
    float s = v;
    #pragma unroll
    for (int o = 16; o > 0; o >>= 1) s += __shfl_xor(s, o, 32);
    if (ln == 0) red[wid] = s;
    __syncthreads();
    float mu = (red[0] + red[1] + red[2] + red[3]) * (1.0f / HID);
    __syncthreads();
    float dv = v - mu, s2 = dv * dv;
    #pragma unroll
    for (int o = 16; o > 0; o >>= 1) s2 += __shfl_xor(s2, o, 32);
    if (ln == 0) red[wid] = s2;
    __syncthreads();
    float var = (red[0] + red[1] + red[2] + red[3]) * (1.0f / HID);
    float xn = dv * rsqrtf(var + 1e-5f) * g[c] + be[c];
    t_bf[(size_t)n * HID + c] = f2bf(silu(xn));
}

// ---------------------------------------------------------------------------
// LayerNorm: h[N,128] f32 -> xn bf16, one wave per node (4 elems/lane)
// ---------------------------------------------------------------------------
__global__ void k_ln(const float* __restrict__ h, const float* __restrict__ g,
                     const float* __restrict__ b, unsigned short* __restrict__ xn,
                     int N) {
    int wave = threadIdx.x >> 5, lane = threadIdx.x & 31;
    int n = blockIdx.x * 8 + wave;
    if (n >= N) return;
    float4 hv = ((const float4*)(h + (size_t)n * HID))[lane];
    float s = hv.x + hv.y + hv.z + hv.w;
    #pragma unroll
    for (int o = 16; o > 0; o >>= 1) s += __shfl_xor(s, o, 32);
    float mu = s * (1.0f / HID);
    float d0 = hv.x - mu, d1 = hv.y - mu, d2 = hv.z - mu, d3 = hv.w - mu;
    float s2 = d0 * d0 + d1 * d1 + d2 * d2 + d3 * d3;
    #pragma unroll
    for (int o = 16; o > 0; o >>= 1) s2 += __shfl_xor(s2, o, 32);
    float rs = rsqrtf(s2 * (1.0f / HID) + 1e-5f);
    int c = lane * 4;
    uint2 u;
    u.x = pk2bf(d0 * rs * g[c + 0] + b[c + 0], d1 * rs * g[c + 1] + b[c + 1]);
    u.y = pk2bf(d2 * rs * g[c + 2] + b[c + 2], d3 * rs * g[c + 3] + b[c + 3]);
    ((uint2*)(xn + (size_t)n * HID))[lane] = u;
}

// ---------------------------------------------------------------------------
// WMMA GEMM: C[M,Nc] = act(A[M,K] @ Wt_bf16[Nc,K] + bias) (* row_scale)
// A is bf16, or f32 packed to bf16 in-register when AF32 (skips a cvt pass).
// Weights staged to LDS via the Tensor Data Mover (one DMA per block).
// ---------------------------------------------------------------------------
template <int K, int NC, int MODE, bool AF32 = false>
__global__ __launch_bounds__(256) void k_gemm(
    const void* __restrict__ Ap, const unsigned short* __restrict__ Wt,
    const float* __restrict__ bias, float* __restrict__ of32,
    unsigned short* __restrict__ ob16, const float* __restrict__ row_scale,
    int M) {
    constexpr int KSTEPS = K / 32;
    constexpr int NTILES = NC / 16;
    __shared__ unsigned short wlds[NC * K];

    int tid = threadIdx.x;
#ifdef HAVE_TDM
    if (tid < 32) {  // wave 0 issues one TDM DMA for the whole block
        tdm_load_to_lds(Wt, (unsigned)(size_t)&wlds[0], NC * K * 2);
        __builtin_amdgcn_s_wait_tensorcnt(0);
    }
#else
    {
        const uint4* src = (const uint4*)Wt;
        uint4* dst = (uint4*)wlds;
        #pragma unroll 2
        for (int i = tid; i < (NC * K) / 8; i += 256) dst[i] = src[i];
    }
#endif
    __syncthreads();

    int wave = tid >> 5, lane = tid & 31;
    int half = lane >> 4, l16 = lane & 15;
    long rowbase = (long)blockIdx.x * 128 + wave * 16;

    // A fragments for all K-steps (lane holds row m = rowbase + l16)
    long mr = rowbase + l16;
    if (mr >= M) mr = M - 1;
    ABu afr[KSTEPS];
    if constexpr (AF32) {
        const float4* ar = (const float4*)((const float*)Ap + (size_t)mr * K);
        #pragma unroll
        for (int kt = 0; kt < KSTEPS; ++kt) {
            float4 f0 = ar[(kt * 32 + 8 * half) >> 2];
            float4 f1 = ar[((kt * 32 + 8 * half) >> 2) + 1];
            float4 f2 = ar[(kt * 32 + 16 + 8 * half) >> 2];
            float4 f3 = ar[((kt * 32 + 16 + 8 * half) >> 2) + 1];
            afr[kt].u4[0] = make_uint4(pk2bf(f0.x, f0.y), pk2bf(f0.z, f0.w),
                                       pk2bf(f1.x, f1.y), pk2bf(f1.z, f1.w));
            afr[kt].u4[1] = make_uint4(pk2bf(f2.x, f2.y), pk2bf(f2.z, f2.w),
                                       pk2bf(f3.x, f3.y), pk2bf(f3.z, f3.w));
        }
    } else {
        const uint4* arow = (const uint4*)((const unsigned short*)Ap + (size_t)mr * K);
        #pragma unroll
        for (int kt = 0; kt < KSTEPS; ++kt) {
            afr[kt].u4[0] = arow[(kt * 32 + 8 * half) >> 3];
            afr[kt].u4[1] = arow[(kt * 32 + 16 + 8 * half) >> 3];
        }
    }

    bool fulltile = (rowbase + 16 <= (long)M);

    #pragma unroll
    for (int nt = 0; nt < NTILES; ++nt) {
        int n = nt * 16 + l16;
        float bv = bias[n];
        v8f acc;
        #pragma unroll
        for (int r = 0; r < 8; ++r) acc[r] = bv;

        const uint4* brow = (const uint4*)(wlds + (size_t)n * K);
        #pragma unroll
        for (int kt = 0; kt < KSTEPS; ++kt) {
            ABu bfr;  // B lane n holds K chunks [kt*32 + 8*half, +8) and [kt*32 + 16 + 8*half, +8)
            bfr.u4[0] = brow[(kt * 32 + 8 * half) >> 3];
            bfr.u4[1] = brow[(kt * 32 + 16 + 8 * half) >> 3];
            acc = __builtin_amdgcn_wmma_f32_16x16x32_bf16(
                false, afr[kt].f, false, bfr.f, (short)0, acc, false, false);
        }

        #pragma unroll
        for (int r = 0; r < 8; ++r) {
            long gm = rowbase + 8 * half + r;
            if (fulltile || gm < M) {
                float val = acc[r];
                if (MODE & GM_SILU) val = silu(val);
                if (MODE & GM_RS)   val *= row_scale[gm];
                size_t o = (size_t)gm * NC + n;
                if (MODE & GM_WF32) { if (MODE & GM_ACC) of32[o] += val; else of32[o] = val; }
                if (MODE & GM_WB16) ob16[o] = f2bf(val);
            }
        }
    }
}

// ---------------------------------------------------------------------------
// Fused edge MLP: wphys = (silu(rbf@W1 + b1) @ W2 + b2) * cut
// Phase 1 (K=32) per wave -> silu -> 16x128 bf16 tile in per-wave LDS
// (same-wave LDS RAW is in-order; no barrier). Phase 2 (K=128) from LDS.
// Avoids the 2x153.6 MB/layer hpre round trip through HBM.
// LDS: 32 KB W2 (TDM) + 32 KB h-tiles = 64 KB. W1 B-frags come from global
// (8 KB, L2-resident across all 4688 blocks).
// ---------------------------------------------------------------------------
__global__ __launch_bounds__(256) void k_edge_mlp(
    const unsigned short* __restrict__ rbf,   // [E,32] bf16
    const unsigned short* __restrict__ W1t,   // [128][32] bf16
    const float* __restrict__ b1,
    const unsigned short* __restrict__ W2t,   // [128][128] bf16
    const float* __restrict__ b2,
    const float* __restrict__ cut,
    unsigned short* __restrict__ wphys, int E) {
    __shared__ unsigned short w2lds[HID * HID];    // 32 KB
    __shared__ unsigned short hlds[8 * 16 * HID];  // 32 KB

    int tid = threadIdx.x;
#ifdef HAVE_TDM
    if (tid < 32) {
        tdm_load_to_lds(W2t, (unsigned)(size_t)&w2lds[0], HID * HID * 2);
        __builtin_amdgcn_s_wait_tensorcnt(0);
    }
#else
    {
        const uint4* src = (const uint4*)W2t;
        uint4* dst = (uint4*)w2lds;
        #pragma unroll 2
        for (int i = tid; i < (HID * HID) / 8; i += 256) dst[i] = src[i];
    }
#endif
    __syncthreads();

    int wave = tid >> 5, lane = tid & 31;
    int half = lane >> 4, l16 = lane & 15;
    long rowbase = (long)blockIdx.x * 128 + wave * 16;
    long mr = rowbase + l16;
    if (mr >= E) mr = E - 1;

    // Phase 1: K=32, A from rbf
    const uint4* arow = (const uint4*)(rbf + (size_t)mr * NRBF);
    ABu a0;
    a0.u4[0] = arow[(8 * half) >> 3];
    a0.u4[1] = arow[(16 + 8 * half) >> 3];

    unsigned short* hrow = hlds + (size_t)wave * 16 * HID;  // wave-private 16x128
    #pragma unroll
    for (int nt = 0; nt < 8; ++nt) {
        int n = nt * 16 + l16;
        float bv = b1[n];
        v8f acc;
        #pragma unroll
        for (int r = 0; r < 8; ++r) acc[r] = bv;
        const uint4* brow = (const uint4*)(W1t + (size_t)n * NRBF);
        ABu bfr;
        bfr.u4[0] = brow[(8 * half) >> 3];
        bfr.u4[1] = brow[(16 + 8 * half) >> 3];
        acc = __builtin_amdgcn_wmma_f32_16x16x32_bf16(
            false, a0.f, false, bfr.f, (short)0, acc, false, false);
        #pragma unroll
        for (int r = 0; r < 8; ++r)
            hrow[(8 * half + r) * HID + n] = f2bf(silu(acc[r]));  // C elem (r,half)->row
    }

    // Phase 2: K=128, A re-fragmented from the per-wave LDS tile
    const uint4* arow2 = (const uint4*)(hrow + (size_t)l16 * HID);
    ABu afr[4];
    #pragma unroll
    for (int kt = 0; kt < 4; ++kt) {
        afr[kt].u4[0] = arow2[(kt * 32 + 8 * half) >> 3];
        afr[kt].u4[1] = arow2[(kt * 32 + 16 + 8 * half) >> 3];
    }

    bool fulltile = (rowbase + 16 <= (long)E);
    #pragma unroll
    for (int nt = 0; nt < 8; ++nt) {
        int n = nt * 16 + l16;
        float bv = b2[n];
        v8f acc;
        #pragma unroll
        for (int r = 0; r < 8; ++r) acc[r] = bv;
        const uint4* brow = (const uint4*)(w2lds + (size_t)n * HID);
        #pragma unroll
        for (int kt = 0; kt < 4; ++kt) {
            ABu bfr;
            bfr.u4[0] = brow[(kt * 32 + 8 * half) >> 3];
            bfr.u4[1] = brow[(kt * 32 + 16 + 8 * half) >> 3];
            acc = __builtin_amdgcn_wmma_f32_16x16x32_bf16(
                false, afr[kt].f, false, bfr.f, (short)0, acc, false, false);
        }
        #pragma unroll
        for (int r = 0; r < 8; ++r) {
            long gm = rowbase + 8 * half + r;
            if (fulltile || gm < E)
                wphys[(size_t)gm * HID + n] = f2bf(acc[r] * cut[gm]);
        }
    }
}

// ---------------------------------------------------------------------------
// Edge messages + softmax-over-heads + atomic scatter. One wave per edge,
// each lane owns 4 contiguous channels (lane/8 = head).
// ---------------------------------------------------------------------------
__global__ void k_msg(const int* __restrict__ ei, const unsigned short* __restrict__ xn,
                      const unsigned short* __restrict__ wp,
                      const unsigned short* __restrict__ Qb,
                      const unsigned short* __restrict__ Kb,
                      const unsigned short* __restrict__ Vb,
                      const float* __restrict__ cut, const float* __restrict__ gate_arr,
                      int l, float* __restrict__ aggr, int E) {
    int wave = threadIdx.x >> 5, lane = threadIdx.x & 31;
    long e = (long)blockIdx.x * 8 + wave;
    if (e >= E) return;
    int row = ei[e], col = ei[(size_t)E + e];
    float cv = cut[e];
    float gate = gate_arr[l];

    uint2 qx = ((const uint2*)(Qb + (size_t)row * HID))[lane];
    uint2 kx = ((const uint2*)(Kb + (size_t)col * HID))[lane];
    uint2 vx = ((const uint2*)(Vb + (size_t)col * HID))[lane];
    uint2 xc = ((const uint2*)(xn + (size_t)col * HID))[lane];
    uint2 wx = ((const uint2*)(wp + (size_t)e * HID))[lane];

    float q0, q1, q2, q3, k0, k1, k2, k3, v0, v1, v2, v3, x0, x1, x2, x3, w0, w1, w2, w3;
    up2(qx.x, q0, q1); up2(qx.y, q2, q3);
    up2(kx.x, k0, k1); up2(kx.y, k2, k3);
    up2(vx.x, v0, v1); up2(vx.y, v2, v3);
    up2(xc.x, x0, x1); up2(xc.y, x2, x3);
    up2(wx.x, w0, w1); up2(wx.y, w2, w3);

    float dp = q0 * k0 + q1 * k1 + q2 * k2 + q3 * k3;
    dp += __shfl_xor(dp, 1, 32);
    dp += __shfl_xor(dp, 2, 32);
    dp += __shfl_xor(dp, 4, 32);                    // per-head dot (8 lanes)
    float sc = dp * 0.1767766953f * cv;             // 1/sqrt(32) * cut
    float s0 = __shfl(sc, 0, 32), s1 = __shfl(sc, 8, 32);
    float s2 = __shfl(sc, 16, 32), s3 = __shfl(sc, 24, 32);
    float mx = fmaxf(fmaxf(s0, s1), fmaxf(s2, s3));
    float den = __expf(s0 - mx) + __expf(s1 - mx) + __expf(s2 - mx) + __expf(s3 - mx);
    float ga = gate * __expf(sc - mx) * __builtin_amdgcn_rcpf(den);

    float* dst = aggr + (size_t)row * HID + lane * 4;
    atomicAdd(dst + 0, x0 * w0 + ga * v0);
    atomicAdd(dst + 1, x1 * w1 + ga * v1);
    atomicAdd(dst + 2, x2 * w2 + ga * v2);
    atomicAdd(dst + 3, x3 * w3 + ga * v3);
}

// head: out[n] = th[n,0:64] . h_w2 + h_b2
__global__ void k_head(const unsigned short* __restrict__ th, const float* __restrict__ w2,
                       const float* __restrict__ b2, float* __restrict__ out, int N) {
    int n = blockIdx.x * 256 + threadIdx.x;
    if (n >= N) return;
    float a = b2[0];
    const unsigned short* r = th + (size_t)n * 64;
    #pragma unroll
    for (int j = 0; j < 64; ++j) a += bf2f(r[j]) * w2[j];
    out[n] = a;
}

// ---------------------------------------------------------------------------
extern "C" void kernel_launch(void* const* d_in, const int* in_sizes, int n_in,
                              void* d_out, int out_size, void* d_ws, size_t ws_size,
                              hipStream_t stream) {
    const float* x    = (const float*)d_in[0];
    const int*   ei   = (const int*)d_in[1];
    const float* ea   = (const float*)d_in[2];
    const float* e_w1 = (const float*)d_in[3];
    const float* e_b1 = (const float*)d_in[4];
    const float* e_g1 = (const float*)d_in[5];
    const float* e_be1= (const float*)d_in[6];
    const float* e_w2 = (const float*)d_in[7];
    const float* e_b2 = (const float*)d_in[8];
    const float* rbf_w1 = (const float*)d_in[9];
    const float* rbf_b1 = (const float*)d_in[10];
    const float* rbf_w2 = (const float*)d_in[11];
    const float* rbf_b2 = (const float*)d_in[12];
    const float* ng   = (const float*)d_in[13];
    const float* nb   = (const float*)d_in[14];
    const float* qw   = (const float*)d_in[15];
    const float* qbb  = (const float*)d_in[16];
    const float* kw   = (const float*)d_in[17];
    const float* kbb  = (const float*)d_in[18];
    const float* vw   = (const float*)d_in[19];
    const float* vbb  = (const float*)d_in[20];
    const float* ow   = (const float*)d_in[21];
    const float* obb  = (const float*)d_in[22];
    const float* gate = (const float*)d_in[23];
    const float* h_g  = (const float*)d_in[24];
    const float* h_b  = (const float*)d_in[25];
    const float* h_w1 = (const float*)d_in[26];
    const float* h_b1 = (const float*)d_in[27];
    const float* h_w2 = (const float*)d_in[28];
    const float* h_b2 = (const float*)d_in[29];
    (void)n_in; (void)ws_size; (void)out_size;

    const int N = in_sizes[0] / 5;
    const int E = in_sizes[2];

    size_t off = 0;
    auto alloc = [&](size_t bytes) -> char* {
        char* p = (char*)d_ws + off;
        off = (off + bytes + 255) & ~(size_t)255;
        return p;
    };
    typedef unsigned short u16;
    u16* wt_ew2  = (u16*)alloc((size_t)HID * HID * 2);
    u16* wt_rbf1 = (u16*)alloc((size_t)NLAYER * NRBF * HID * 2);
    u16* wt_rbf2 = (u16*)alloc((size_t)NLAYER * HID * HID * 2);
    u16* wt_q    = (u16*)alloc((size_t)NLAYER * HID * HID * 2);
    u16* wt_k    = (u16*)alloc((size_t)NLAYER * HID * HID * 2);
    u16* wt_v    = (u16*)alloc((size_t)NLAYER * HID * HID * 2);
    u16* wt_o    = (u16*)alloc((size_t)NLAYER * HID * HID * 2);
    u16* wt_h1   = (u16*)alloc((size_t)64 * HID * 2);
    float* h     = (float*)alloc((size_t)N * HID * 4);
    u16* xn_bf   = (u16*)alloc((size_t)N * HID * 2);
    u16* q_bf    = (u16*)alloc((size_t)N * HID * 2);
    u16* k_bf    = (u16*)alloc((size_t)N * HID * 2);
    u16* v_bf    = (u16*)alloc((size_t)N * HID * 2);
    u16* rbf_bf  = (u16*)alloc((size_t)E * NRBF * 2);
    float* cutw  = (float*)alloc((size_t)E * 4);
    u16* wphys   = (u16*)alloc((size_t)E * HID * 2);
    float* aggr  = (float*)alloc((size_t)N * HID * 4);
    u16* th_bf   = q_bf;  // reuse after attention is done

    const int T128 = (HID * HID + 255) / 256;
    const int T32  = (NRBF * HID + 255) / 256;
    const int T64  = (64 * HID + 255) / 256;

    // --- weight prep (transpose + bf16) ---
    k_tw<<<T128, 256, 0, stream>>>(e_w2, wt_ew2, HID, HID);
    for (int l = 0; l < NLAYER; ++l) {
        k_tw<<<T32, 256, 0, stream>>>(rbf_w1 + (size_t)l * NRBF * HID, wt_rbf1 + (size_t)l * NRBF * HID, NRBF, HID);
        k_tw<<<T128, 256, 0, stream>>>(rbf_w2 + (size_t)l * HID * HID, wt_rbf2 + (size_t)l * HID * HID, HID, HID);
        k_tw<<<T128, 256, 0, stream>>>(qw + (size_t)l * HID * HID, wt_q + (size_t)l * HID * HID, HID, HID);
        k_tw<<<T128, 256, 0, stream>>>(kw + (size_t)l * HID * HID, wt_k + (size_t)l * HID * HID, HID, HID);
        k_tw<<<T128, 256, 0, stream>>>(vw + (size_t)l * HID * HID, wt_v + (size_t)l * HID * HID, HID, HID);
        k_tw<<<T128, 256, 0, stream>>>(ow + (size_t)l * HID * HID, wt_o + (size_t)l * HID * HID, HID, HID);
    }
    k_tw<<<T64, 256, 0, stream>>>(h_w1, wt_h1, HID, 64);

    // --- edge basis ---
    k_edge_rbf<<<(E + 255) / 256, 256, 0, stream>>>(ea, rbf_bf, cutw, E);

    // --- embed: t = silu(LN(x@W1+b)); h = t@W2 + b2 ---
    k_embed<<<N, HID, 0, stream>>>(x, e_w1, e_b1, e_g1, e_be1, xn_bf, N);
    const int GN = (N + 127) / 128, GE = (E + 127) / 128;
    k_gemm<HID, HID, GM_WF32><<<GN, 256, 0, stream>>>(xn_bf, wt_ew2, e_b2, h, nullptr, nullptr, N);

    for (int l = 0; l < NLAYER; ++l) {
        k_ln<<<(N + 7) / 8, 256, 0, stream>>>(h, ng + l * HID, nb + l * HID, xn_bf, N);
        // Q/K/V projections
        k_gemm<HID, HID, GM_WB16><<<GN, 256, 0, stream>>>(xn_bf, wt_q + (size_t)l * HID * HID, qbb + l * HID, nullptr, q_bf, nullptr, N);
        k_gemm<HID, HID, GM_WB16><<<GN, 256, 0, stream>>>(xn_bf, wt_k + (size_t)l * HID * HID, kbb + l * HID, nullptr, k_bf, nullptr, N);
        k_gemm<HID, HID, GM_WB16><<<GN, 256, 0, stream>>>(xn_bf, wt_v + (size_t)l * HID * HID, vbb + l * HID, nullptr, v_bf, nullptr, N);
        // fused edge physics MLP: wphys = (silu(rbf@W1+b1)@W2+b2)*cut
        k_edge_mlp<<<GE, 256, 0, stream>>>(rbf_bf, wt_rbf1 + (size_t)l * NRBF * HID, rbf_b1 + l * HID,
                                           wt_rbf2 + (size_t)l * HID * HID, rbf_b2 + l * HID, cutw, wphys, E);
        // scatter messages
        (void)hipMemsetAsync(aggr, 0, (size_t)N * HID * 4, stream);
        k_msg<<<(E + 7) / 8, 256, 0, stream>>>(ei, xn_bf, wphys, q_bf, k_bf, v_bf, cutw, gate, l, aggr, E);
        // h += aggr @ ow + ob   (aggr is f32; packed to bf16 in-register)
        k_gemm<HID, HID, GM_WF32 | GM_ACC, true><<<GN, 256, 0, stream>>>(aggr, wt_o + (size_t)l * HID * HID, obb + l * HID, h, nullptr, nullptr, N);
    }

    // --- head ---
    k_ln<<<(N + 7) / 8, 256, 0, stream>>>(h, h_g, h_b, xn_bf, N);
    k_gemm<HID, 64, GM_WB16 | GM_SILU><<<GN, 256, 0, stream>>>(xn_bf, wt_h1, h_b1, nullptr, th_bf, nullptr, N);
    k_head<<<(N + 255) / 256, 256, 0, stream>>>(th_bf, h_w2, h_b2, (float*)d_out, N);
}
